// Encoder_62723702391359
// MI455X (gfx1250) — matrix operand order repaired and checked by reading, not compiled, not run on hardware
//
#include <hip/hip_runtime.h>
#include <hip/hip_fp16.h>
#include <cmath>

typedef __attribute__((ext_vector_type(16))) _Float16 v16h;
typedef __attribute__((ext_vector_type(8)))  _Float16 v8h;
typedef __attribute__((ext_vector_type(2)))  _Float16 v2h;
typedef __attribute__((ext_vector_type(8)))  float    v8f;

#define NN    50000
#define NE    1600000
#define ETOT  (NE + NN)      // edges + self loops
#define INDIM 512
#define HID   128
#define OUT2  64             // OUT_DIM*2 (layer-2 channel count)
#define NEG   0.2f

// ---------------------------------------------------------------- utilities
__global__ void fill_kernel(float* __restrict__ p, float v, int n) {
  int i = blockIdx.x * blockDim.x + threadIdx.x;
  if (i < n) p[i] = v;
}

__device__ inline void atomic_max_f32(float* addr, float val) {
  // signed-int max for non-negative floats, unsigned-int min for negative:
  // correct total-order max given -inf initialization.
  if (val >= 0.0f) atomicMax((int*)addr, __float_as_int(val));
  else             atomicMin((unsigned int*)addr, __float_as_uint(val));
}

// ------------------------------------------------- dual-B WMMA GEMM (f16 MACs)
// C_l = A*B_l + bias_l ; C_r = A*B_r + bias_r      A:[M,K] f32, B:[K,N] f32
// Block: 128 threads = 4 waves; block tile 64(M) x 64(N); K stepped by 32.
// Wave w owns N-strip w (16 cols) and iterates 4 M-tiles, reusing B fragments.
// LDS: A tile row-major (per-lane A fragment = two contiguous 16B runs),
//      B tiles transposed (per-lane B fragment = two contiguous 16B runs),
//      both padded to stride 40 halves (80B: keeps 16B alignment, spreads banks).
template<int K, int N>
__global__ __launch_bounds__(128)
void wmma_dual_gemm(const float* __restrict__ A,
                    const float* __restrict__ Bl, const float* __restrict__ Br,
                    const float* __restrict__ bl, const float* __restrict__ br,
                    float* __restrict__ Cl, float* __restrict__ Cr)
{
  constexpr int LDT = 40;                 // padded LDS stride in f16 units
  __shared__ _Float16 sA [64 * LDT];      // [m][k]   (row-major)
  __shared__ _Float16 sBl[64 * LDT];      // [n][k]   (transposed)
  __shared__ _Float16 sBr[64 * LDT];

  const int tid  = threadIdx.x;
  const int lane = tid & 31;
  const int wave = tid >> 5;
  const int m0   = blockIdx.x * 64;
  const int n0   = blockIdx.y * 64;

  v8f aL[4] = {{}, {}, {}, {}};
  v8f aR[4] = {{}, {}, {}, {}};

  const int mrow = lane & 15;
  const int hh   = lane >> 4;             // K-phase half of the wave
  const int ncol = (wave << 4) + (lane & 15);
  const int kb   = hh << 4;               // B fragment K base (0 or 16)

  for (int k0 = 0; k0 < K; k0 += 32) {
    // ---- stage A tile 64x32: float2 global load -> packed f16 b32 LDS store
#pragma unroll
    for (int t = 0; t < 8; ++t) {
      int i  = tid + t * 128;
      int r  = i >> 4;                    // 0..63
      int c2 = (i & 15) << 1;             // 0,2,..,30
      int row = m0 + r; if (row > NN - 1) row = NN - 1;   // clamp (edge block)
      float2 f = *(const float2*)(A + (size_t)row * K + k0 + c2);
      *(v2h*)(sA + r * LDT + c2) = (v2h){ (_Float16)f.x, (_Float16)f.y };
    }
    // ---- stage B tiles 32x64, transposed: pack rows (r2, r2+1) per column
#pragma unroll
    for (int t = 0; t < 8; ++t) {
      int i  = tid + t * 128;
      int c  = i & 63;                    // column (n offset)
      int r2 = (i >> 6) << 1;             // 0,2,..,30
      float l0 = Bl[(size_t)(k0 + r2)     * N + n0 + c];
      float l1 = Bl[(size_t)(k0 + r2 + 1) * N + n0 + c];
      *(v2h*)(sBl + c * LDT + r2) = (v2h){ (_Float16)l0, (_Float16)l1 };
      float r0 = Br[(size_t)(k0 + r2)     * N + n0 + c];
      float r1 = Br[(size_t)(k0 + r2 + 1) * N + n0 + c];
      *(v2h*)(sBr + c * LDT + r2) = (v2h){ (_Float16)r0, (_Float16)r1 };
    }
    if (k0 + 32 < K)                      // global_prefetch_b8 of next A tile
      __builtin_prefetch(A + (size_t)(m0 + (tid & 63)) * K + k0 + 32, 0, 0);
    __syncthreads();

    // ---- B fragments: k = kb .. kb+15 at fixed column -> 2x ds_load_b128
    v8h b0, b1;
    b0 = *(const v8h*)(sBl + ncol * LDT + kb);
    b1 = *(const v8h*)(sBl + ncol * LDT + kb + 8);
    v16h bfl = __builtin_shufflevector(b0, b1, 0,1,2,3,4,5,6,7,8,9,10,11,12,13,14,15);
    b0 = *(const v8h*)(sBr + ncol * LDT + kb);
    b1 = *(const v8h*)(sBr + ncol * LDT + kb + 8);
    v16h bfr = __builtin_shufflevector(b0, b1, 0,1,2,3,4,5,6,7,8,9,10,11,12,13,14,15);

    // ---- 4 M-tiles per wave, A fragment = two contiguous 16B runs
#pragma unroll
    for (int mt = 0; mt < 4; ++mt) {
      const _Float16* pa = sA + (mt * 16 + mrow) * LDT + hh * 16;
      v8h a0 = *(const v8h*)(pa);         // k = hh*16 + 0..7
      v8h a1 = *(const v8h*)(pa + 8);     // k = hh*16 + 8..15
      v16h af = __builtin_shufflevector(a0, a1, 0,1,2,3,4,5,6,7,8,9,10,11,12,13,14,15);
      aL[mt] = __builtin_amdgcn_wmma_f32_16x16x32_f16(false, af, false, bfl,
                                                      (short)0, aL[mt], false, false);
      aR[mt] = __builtin_amdgcn_wmma_f32_16x16x32_f16(false, af, false, bfr,
                                                      (short)0, aR[mt], false, false);
    }
    __syncthreads();
  }

  // ---- epilogue: C/D layout -> vgpr r holds (M = 8*(lane>>4)+r, N = lane&15)
  const int n = n0 + (wave << 4) + (lane & 15);
  const float addL = bl[n];
  const float addR = br[n];
#pragma unroll
  for (int mt = 0; mt < 4; ++mt) {
    int mb = m0 + mt * 16 + ((lane >> 4) << 3);
#pragma unroll
    for (int r = 0; r < 8; ++r) {
      int row = mb + r;
      if (row < NN) {
        Cl[(size_t)row * N + n] = aL[mt][r] + addL;
        Cr[(size_t)row * N + n] = aR[mt][r] + addR;
      }
    }
  }
}

// ------------------------------------------- edge pass 1: logits + segment max
// one wave (32 lanes) per edge; lane owns channels 4*lane..4*lane+3 (b128 loads)
template<int C>
__global__ __launch_bounds__(256)
void edge_logits(const float* __restrict__ xl, const float* __restrict__ xr,
                 const int* __restrict__ src, const int* __restrict__ dst,
                 const float* __restrict__ att,
                 float* __restrict__ logits, float* __restrict__ mmax)
{
  const int e = blockIdx.x * 8 + (threadIdx.x >> 5);
  if (e >= ETOT) return;
  const int lane = threadIdx.x & 31;
  int sN, dN;
  if (e < NE) { sN = src[e]; dN = dst[e]; } else { sN = dN = e - NE; }

  float acc = 0.0f;
  if constexpr (C == 128) {
    const float4 a = *(const float4*)(xl + (size_t)sN * C + lane * 4);
    const float4 b = *(const float4*)(xr + (size_t)dN * C + lane * 4);
    const float4 w = *(const float4*)(att + lane * 4);
    float v;
    v = a.x + b.x; v = (v > 0.0f) ? v : NEG * v; acc += v * w.x;
    v = a.y + b.y; v = (v > 0.0f) ? v : NEG * v; acc += v * w.y;
    v = a.z + b.z; v = (v > 0.0f) ? v : NEG * v; acc += v * w.z;
    v = a.w + b.w; v = (v > 0.0f) ? v : NEG * v; acc += v * w.w;
  } else {                                // C == 64
    const float2 a = *(const float2*)(xl + (size_t)sN * C + lane * 2);
    const float2 b = *(const float2*)(xr + (size_t)dN * C + lane * 2);
    const float2 w = *(const float2*)(att + lane * 2);
    float v;
    v = a.x + b.x; v = (v > 0.0f) ? v : NEG * v; acc += v * w.x;
    v = a.y + b.y; v = (v > 0.0f) ? v : NEG * v; acc += v * w.y;
  }
#pragma unroll
  for (int off = 16; off; off >>= 1) acc += __shfl_xor(acc, off, 32);

  if (lane == 0) {
    logits[e] = acc;
    atomic_max_f32(&mmax[dN], acc);
  }
}

// ------------- edge pass 2: a = exp(logit - m[dst]); scatter a and a*xl[src]
// (final division by s happens node-wise: sum(a*x)/sum(a) == softmax aggregate)
template<int C>
__global__ __launch_bounds__(256)
void edge_scatter(const float* __restrict__ xl,
                  const int* __restrict__ src, const int* __restrict__ dst,
                  const float* __restrict__ logits, const float* __restrict__ mmax,
                  float* __restrict__ ssum, float* __restrict__ acc)
{
  const int e = blockIdx.x * 8 + (threadIdx.x >> 5);
  if (e >= ETOT) return;
  const int lane = threadIdx.x & 31;
  int sN, dN;
  if (e < NE) { sN = src[e]; dN = dst[e]; } else { sN = dN = e - NE; }

  const float a = __expf(logits[e] - mmax[dN]);
  if (lane == 0) atomicAdd(&ssum[dN], a);
  if constexpr (C == 128) {
    const float4 xv = *(const float4*)(xl + (size_t)sN * C + lane * 4);
    float* base = acc + (size_t)dN * C + lane * 4;
    atomicAdd(base + 0, a * xv.x);
    atomicAdd(base + 1, a * xv.y);
    atomicAdd(base + 2, a * xv.z);
    atomicAdd(base + 3, a * xv.w);
  } else {                                // C == 64
    const float2 xv = *(const float2*)(xl + (size_t)sN * C + lane * 2);
    float* base = acc + (size_t)dN * C + lane * 2;
    atomicAdd(base + 0, a * xv.x);
    atomicAdd(base + 1, a * xv.y);
  }
}

// --------------------------------------- node-wise normalize + bias (+ relu)
__global__ void normalize_kernel(float* __restrict__ acc,
                                 const float* __restrict__ ssum,
                                 const float* __restrict__ bias,
                                 int C, int do_relu)
{
  int i = blockIdx.x * blockDim.x + threadIdx.x;
  if (i >= NN * C) return;
  int node = i / C;
  int c    = i - node * C;
  float v = acc[i] / (ssum[node] + 1e-16f) + bias[c];
  if (do_relu) v = fmaxf(v, 0.0f);
  acc[i] = v;
}

// ---------------------------------------------------------------- launcher
extern "C" void kernel_launch(void* const* d_in, const int* in_sizes, int n_in,
                              void* d_out, int out_size, void* d_ws, size_t ws_size,
                              hipStream_t stream)
{
  const float* x     = (const float*)d_in[0];
  const float* Wl1   = (const float*)d_in[1];
  const float* bl1   = (const float*)d_in[2];
  const float* Wr1   = (const float*)d_in[3];
  const float* br1   = (const float*)d_in[4];
  const float* att1  = (const float*)d_in[5];
  const float* bias1 = (const float*)d_in[6];
  const float* Wl2   = (const float*)d_in[7];
  const float* bl2   = (const float*)d_in[8];
  const float* Wr2   = (const float*)d_in[9];
  const float* br2   = (const float*)d_in[10];
  const float* att2  = (const float*)d_in[11];
  const float* bias2 = (const float*)d_in[12];
  const int*   eidx  = (const int*)d_in[13];
  const int*   srcv  = eidx;        // edge_index[0]
  const int*   dstv  = eidx + NE;   // edge_index[1]
  float* out = (float*)d_out;

  // workspace layout (floats): ~110 MB total
  float* ws   = (float*)d_ws;
  float* xl1  = ws;  ws += (size_t)NN * HID;
  float* xr1  = ws;  ws += (size_t)NN * HID;
  float* h    = ws;  ws += (size_t)NN * HID;   // scatter accumulator -> hidden
  float* xl2  = ws;  ws += (size_t)NN * OUT2;
  float* xr2  = ws;  ws += (size_t)NN * OUT2;
  float* lg   = ws;  ws += (size_t)ETOT;       // per-edge logits
  float* mmax = ws;  ws += (size_t)NN;
  float* ssum = ws;  ws += (size_t)NN;

  const int egrid = (ETOT + 7) / 8;            // 8 waves (edges) per 256-thr block

  // ---- layer 1 ----
  fill_kernel<<<(NN + 255) / 256, 256, 0, stream>>>(mmax, -INFINITY, NN);
  fill_kernel<<<(NN + 255) / 256, 256, 0, stream>>>(ssum, 0.0f, NN);
  fill_kernel<<<(NN * HID + 255) / 256, 256, 0, stream>>>(h, 0.0f, NN * HID);
  fill_kernel<<<(NN * OUT2 + 255) / 256, 256, 0, stream>>>(out, 0.0f, NN * OUT2);

  dim3 g1((NN + 63) / 64, HID / 64);           // 782 x 2
  wmma_dual_gemm<INDIM, HID><<<g1, 128, 0, stream>>>(x, Wl1, Wr1, bl1, br1, xl1, xr1);

  edge_logits<HID><<<egrid, 256, 0, stream>>>(xl1, xr1, srcv, dstv, att1, lg, mmax);
  edge_scatter<HID><<<egrid, 256, 0, stream>>>(xl1, srcv, dstv, lg, mmax, ssum, h);
  normalize_kernel<<<(NN * HID + 255) / 256, 256, 0, stream>>>(h, ssum, bias1, HID, 1);

  // ---- layer 2 ----
  fill_kernel<<<(NN + 255) / 256, 256, 0, stream>>>(mmax, -INFINITY, NN);
  fill_kernel<<<(NN + 255) / 256, 256, 0, stream>>>(ssum, 0.0f, NN);

  dim3 g2((NN + 63) / 64, OUT2 / 64);          // 782 x 1
  wmma_dual_gemm<HID, OUT2><<<g2, 128, 0, stream>>>(h, Wl2, Wr2, bl2, br2, xl2, xr2);

  edge_logits<OUT2><<<egrid, 256, 0, stream>>>(xl2, xr2, srcv, dstv, att2, lg, mmax);
  edge_scatter<OUT2><<<egrid, 256, 0, stream>>>(xl2, srcv, dstv, lg, mmax, ssum, out);
  normalize_kernel<<<(NN * OUT2 + 255) / 256, 256, 0, stream>>>(out, ssum, bias2, OUT2, 0);
}
